// Block_6004364280078
// MI455X (gfx1250) — matrix operand order repaired and checked by reading, not compiled, not run on hardware
//
#include <hip/hip_runtime.h>
#include <hip/hip_bf16.h>
#include <math.h>

// ---------------- problem constants ----------------
#define NB 4
#define BT 2048
#define CD 1024
#define NH 16
#define HS 64
#define FF 4096
#define MROWS (NB * BT)   // 8192

#define LDS_STR 40        // bf16 elements; 80B row stride = 64B data + 16B pad

// ---------------- CDNA5 WMMA types ----------------
typedef __attribute__((ext_vector_type(16))) __bf16 v16bf;
typedef __attribute__((ext_vector_type(8)))  float  v8f;
typedef __attribute__((ext_vector_type(4)))  unsigned int v4ui;
typedef __attribute__((ext_vector_type(8)))  int v8si;
typedef __attribute__((ext_vector_type(4)))  int v4si;

struct alignas(16) B16 { unsigned int w[4]; };          // 16-byte load/store unit
union Frag16 { B16 u[2]; v16bf v; };                    // one 16x32 bf16 operand / lane

#if defined(__has_builtin)
#if __has_builtin(__builtin_amdgcn_tensor_load_to_lds) && \
    __has_builtin(__builtin_amdgcn_s_wait_tensorcnt)
#define HAVE_TDM 1
#endif
#endif

__device__ __forceinline__ unsigned short f2bf(float f) {
  unsigned int u = __builtin_bit_cast(unsigned int, f);
  u += 0x7FFFu + ((u >> 16) & 1u);                       // round-to-nearest-even
  return (unsigned short)(u >> 16);
}

__device__ __forceinline__ v8f wmma_bf16(const v16bf& a, const v16bf& b, v8f c) {
  return __builtin_amdgcn_wmma_f32_16x16x32_bf16(false, a, false, b, (short)0, c,
                                                 false, false);
}

__device__ __forceinline__ unsigned lds_offset(const void* p) {
  // generic -> LDS address space -> 32-bit LDS byte offset
  return (unsigned)(unsigned long long)
         (__attribute__((address_space(3))) const void*)p;
}

#ifdef HAVE_TDM
// One TDM 2D tile load: tile_d1 rows x tile_d0 elems (2B each) from row-major
// global (row stride = srcK elems) into LDS at ldsbyte, padding 16B after
// every 64B stored (matches LDS_STR=40 bf16 row pitch).
__device__ __forceinline__ void tdm_load_tile_2d(unsigned long long gaddr,
                                                 unsigned ldsbyte, int srcK,
                                                 int tensor_d1, int tile_d0,
                                                 int tile_d1) {
  v4ui g0;
  g0[0] = 1u;                               // count=1, user desc, gather off
  g0[1] = ldsbyte;                          // lds_addr [63:32]
  g0[2] = (unsigned)(gaddr & 0xFFFFFFFFull);        // global_addr [95:64]
  g0[3] = (unsigned)((gaddr >> 32) & 0x1FFFFFFull)  // global_addr [120:96]
          | (2u << 30);                     // type=2 ("image") [127:126]

  v8si g1;
  g1[0] = (int)((1u << 16)                  // data_size = 2 bytes
                | (1u << 20)                // pad_enable
                | (3u << 22)                // pad_interval: 16 DWORDs (64B)
                | (3u << 25));              // pad_amount: 4 DWORDs (16B)
  g1[1] = (int)(((unsigned)srcK & 0xFFFFu) << 16);          // tensor_dim0 lo
  g1[2] = (int)((((unsigned)srcK >> 16) & 0xFFFFu)
                | (((unsigned)tensor_d1 & 0xFFFFu) << 16)); // dim0 hi | dim1 lo
  g1[3] = (int)((((unsigned)tensor_d1 >> 16) & 0xFFFFu)
                | (((unsigned)tile_d0 & 0xFFFFu) << 16));   // dim1 hi | tile_dim0
  g1[4] = (int)((unsigned)tile_d1 & 0xFFFFu);               // tile_dim1 (tile_dim2=0)
  g1[5] = (int)(unsigned)srcK;                              // tensor_dim0_stride lo32
  g1[6] = 0;
  g1[7] = 0;
  v4si gz = {0, 0, 0, 0};
#if __clang_major__ >= 23
  v8si gz8 = {0, 0, 0, 0, 0, 0, 0, 0};
  __builtin_amdgcn_tensor_load_to_lds(g0, g1, gz, gz, gz8, 0);
#else
  __builtin_amdgcn_tensor_load_to_lds(g0, g1, gz, gz, 0);
#endif
}
#endif

// ---------------- f32 [K,N] -> bf16 [N,K] transpose-convert ----------------
// grid (K/32, N/32, batch); block 256 (32x8); LDS-tiled, conflict-padded.
__global__ void __launch_bounds__(256)
cvtT_bf16_kernel(const float* __restrict__ in, unsigned short* __restrict__ out,
                 int K, int N) {
  __shared__ unsigned short tile[32][33];
  const size_t zoff = (size_t)blockIdx.z * (size_t)K * N;
  const int k0 = blockIdx.x * 32, n0 = blockIdx.y * 32;
  const int tx = threadIdx.x & 31, ty = threadIdx.x >> 5;
#pragma unroll
  for (int i = 0; i < 32; i += 8)
    tile[ty + i][tx] = f2bf(in[zoff + (size_t)(k0 + ty + i) * N + n0 + tx]);
  __syncthreads();
#pragma unroll
  for (int i = 0; i < 32; i += 8)
    out[zoff + (size_t)(n0 + ty + i) * K + k0 + tx] = tile[tx][ty + i];
}

// ---------------- LayerNorm (row = 1024 floats) -> bf16 ----------------
__global__ void __launch_bounds__(256)
ln_kernel(const float* __restrict__ x, const float* __restrict__ g,
          const float* __restrict__ be, unsigned short* __restrict__ out) {
  __shared__ float red[256];
  const int tid = threadIdx.x;
  const size_t row = blockIdx.x;
  const float* xr = x + row * CD;
  float v0 = xr[tid], v1 = xr[tid + 256], v2 = xr[tid + 512], v3 = xr[tid + 768];

  red[tid] = v0 + v1 + v2 + v3;
  __syncthreads();
  for (int s = 128; s > 0; s >>= 1) {
    if (tid < s) red[tid] += red[tid + s];
    __syncthreads();
  }
  const float mu = red[0] * (1.0f / CD);
  __syncthreads();

  float d0 = v0 - mu, d1 = v1 - mu, d2 = v2 - mu, d3 = v3 - mu;
  red[tid] = d0 * d0 + d1 * d1 + d2 * d2 + d3 * d3;
  __syncthreads();
  for (int s = 128; s > 0; s >>= 1) {
    if (tid < s) red[tid] += red[tid + s];
    __syncthreads();
  }
  const float inv = rsqrtf(red[0] * (1.0f / CD) + 1e-5f);

  unsigned short* orow = out + row * CD;
  orow[tid]       = f2bf(d0 * inv * g[tid]       + be[tid]);
  orow[tid + 256] = f2bf(d1 * inv * g[tid + 256] + be[tid + 256]);
  orow[tid + 512] = f2bf(d2 * inv * g[tid + 512] + be[tid + 512]);
  orow[tid + 768] = f2bf(d3 * inv * g[tid + 768] + be[tid + 768]);
}

// ---------------- generic bf16 WMMA GEMM (B pre-transposed) ----------------
// C[M,N] = A[M,K](bf16,rm) x Bt[N,K](bf16,rm)^T  (+bias)(relu)(+resid)
// OUTM: 0 = f32 out, 1 = bf16 out,
//       2 = bf16 QKV scatter [B,H,T,hs], 3 = bf16 QKV scatter [B,H,hs,T]
// Block: 256 thr = 8 waves; tile 128(M) x 64(N); waves 4x2, each 32x32.
// A tile staged to LDS by the Tensor Data Mover (padded rows); B fragments
// load directly from global (row of Bt == WMMA B column, K contiguous).
template <bool BIAS, bool RELU, bool RES, int OUTM>
__global__ void __launch_bounds__(256)
gemm_bf16_kernel(const unsigned short* __restrict__ A,
                 const unsigned short* __restrict__ Bt,
                 const float* __restrict__ bias,
                 const float* __restrict__ resid,
                 float* __restrict__ outF,
                 unsigned short* __restrict__ outB,
                 int M, int N, int K) {
  __shared__ alignas(16) unsigned short ldsA[128 * LDS_STR];

  const int tid  = threadIdx.x;
  const int lane = tid & 31;
  const int l16  = lane & 15;
  const int hi   = lane >> 4;            // 0: lanes 0-15, 1: lanes 16-31
  const int wave = tid >> 5;
  const int wm   = (wave >> 1) * 32;     // wave M origin within block tile
  const int wn   = (wave & 1) * 32;      // wave N origin within block tile
  const int m0   = blockIdx.x * 128;
  const int n0   = blockIdx.y * 64;

  if constexpr (OUTM >= 2) Bt += (size_t)blockIdx.z * (size_t)K * N;

#ifdef HAVE_TDM
  const unsigned ldsAbyte = lds_offset(ldsA);
  const unsigned long long gA =
      (unsigned long long)(const void*)(A + (size_t)m0 * K);  // tile row 0, k=0
#endif

  // per-lane B row pointers (rows of Bt = output columns)
  const unsigned short* bp0 = Bt + (size_t)(n0 + wn + l16) * K + hi * 16;
  const unsigned short* bp1 = Bt + (size_t)(n0 + wn + 16 + l16) * K + hi * 16;

  v8f acc[2][2];
  for (int i = 0; i < 2; i++)
    for (int j = 0; j < 2; j++) acc[i][j] = v8f{};

  for (int k0 = 0; k0 < K; k0 += 32) {
    __syncthreads();
#ifdef HAVE_TDM
    if (wave == 0) {  // TDM: 128 rows x 32 bf16, 16B pad per 64B row
      tdm_load_tile_2d(gA + (unsigned)k0 * 2u, ldsAbyte, K, M - m0, 32, 128);
      __builtin_amdgcn_s_wait_tensorcnt((short)0);
    }
#else
    {  // stage A tile 128x32 (row-major, stride LDS_STR)
      const int row = tid >> 1, half = (tid & 1) * 16;
      const unsigned short* src = A + (size_t)(m0 + row) * K + k0 + half;
      B16 t0 = *(const B16*)src;
      B16 t1 = *(const B16*)(src + 8);
      *(B16*)(ldsA + row * LDS_STR + half)     = t0;
      *(B16*)(ldsA + row * LDS_STR + half + 8) = t1;
    }
#endif
    __syncthreads();

    Frag16 af[2], bf[2];
#pragma unroll
    for (int i = 0; i < 2; i++) {  // A frag: lane<16 K{0-15}; lane>=16 K{16-31}
      const unsigned short* p = ldsA + (wm + i * 16 + l16) * LDS_STR;
      af[i].u[0] = *(const B16*)(p + hi * 16);
      af[i].u[1] = *(const B16*)(p + hi * 16 + 8);
    }
    // B frag: lane = column, 16 K-contiguous bf16 per half, straight from global
    bf[0].u[0] = *(const B16*)(bp0 + k0);
    bf[0].u[1] = *(const B16*)(bp0 + k0 + 8);
    bf[1].u[0] = *(const B16*)(bp1 + k0);
    bf[1].u[1] = *(const B16*)(bp1 + k0 + 8);
    if (k0 + 32 < K) {
      __builtin_prefetch(bp0 + k0 + 32, 0, 1);
      __builtin_prefetch(bp1 + k0 + 32, 0, 1);
    }
#pragma unroll
    for (int i = 0; i < 2; i++)
#pragma unroll
      for (int j = 0; j < 2; j++) acc[i][j] = wmma_bf16(af[i].v, bf[j].v, acc[i][j]);
  }

  // epilogue: C frag VGPR r -> row r (lanes 0-15) / r+8 (lanes 16-31), col = lane%16
#pragma unroll
  for (int i = 0; i < 2; i++)
#pragma unroll
    for (int j = 0; j < 2; j++) {
      const int n = n0 + wn + j * 16 + l16;
#pragma unroll
      for (int r = 0; r < 8; r++) {
        const int m = m0 + wm + i * 16 + r + hi * 8;
        float v = acc[i][j][r];
        if constexpr (BIAS) v += bias[n];
        if constexpr (RELU) v = fmaxf(v, 0.0f);
        if constexpr (RES)  v += resid[(size_t)m * N + n];
        if constexpr (OUTM == 0) {
          outF[(size_t)m * N + n] = v;
        } else if constexpr (OUTM == 1) {
          outB[(size_t)m * N + n] = f2bf(v);
        } else if constexpr (OUTM == 2) {
          const int b = m / BT, t = m % BT;   // scatter into [B,H,T,hs]
          outB[(((size_t)(b * NH + blockIdx.z) * BT) + t) * HS + n] = f2bf(v);
        } else {
          const int b = m / BT, t = m % BT;   // scatter into [B,H,hs,T]
          outB[(((size_t)(b * NH + blockIdx.z) * HS) + n) * BT + t] = f2bf(v);
        }
      }
    }
}

// ---------------- flash attention: 1 wave per 16 query rows ----------------
// q,k: [B,H,T,hs] bf16; vT: [B,H,hs,T] bf16 (so P*V B-fragments are contiguous)
__global__ void __launch_bounds__(32)
attn_kernel(const unsigned short* __restrict__ q,
            const unsigned short* __restrict__ k,
            const unsigned short* __restrict__ vT,
            unsigned short* __restrict__ attnout) {
  __shared__ alignas(16) unsigned short ldsP[16 * 32];  // P tile, 16x32

  const int lane = threadIdx.x;
  const int l16  = lane & 15;
  const int hi   = lane >> 4;
  const int q0   = blockIdx.x * 16;
  const int head = blockIdx.y;
  const int b    = blockIdx.z;
  const size_t base = ((size_t)(b * NH + head) * BT) * HS;
  const unsigned short* Q  = q + base;
  const unsigned short* Kp = k + base;
  const unsigned short* VT = vT + base;  // same element count, [hs][T] layout

  Frag16 qa[2];  // Q rows q0..q0+15, channels 0..63 as two K=32 A-fragments
  {
    const unsigned short* p = Q + (size_t)(q0 + l16) * HS;
#pragma unroll
    for (int c = 0; c < 2; c++) {
      qa[c].u[0] = *(const B16*)(p + c * 32 + hi * 16);
      qa[c].u[1] = *(const B16*)(p + c * 32 + hi * 16 + 8);
    }
  }

  float rowmax[8], rowsum[8];
  v8f o[4];
#pragma unroll
  for (int r = 0; r < 8; r++) { rowmax[r] = -1e30f; rowsum[r] = 0.0f; }
#pragma unroll
  for (int nt = 0; nt < 4; nt++) o[nt] = v8f{};

  const float scale = 0.125f;  // 1/sqrt(64)

  for (int kv0 = 0; kv0 < q0 + 16; kv0 += 32) {  // kv0 <= q0 always
    float s[2][8];
#pragma unroll
    for (int nh = 0; nh < 2; nh++) {  // S[:, nh*16:+16] = Q x K^T
      const int kvn0 = kv0 + nh * 16;
      v8f sa = v8f{};
#pragma unroll
      for (int c = 0; c < 2; c++) {
        // B-frag of K^T: lane = kv row (column of K^T), channels contiguous
        Frag16 kb;
        const unsigned short* p = Kp + (size_t)(kvn0 + l16) * HS + c * 32 + hi * 16;
        kb.u[0] = *(const B16*)p;
        kb.u[1] = *(const B16*)(p + 8);
        sa = wmma_bf16(qa[c].v, kb.v, sa);
      }
#pragma unroll
      for (int r = 0; r < 8; r++) {
        const int qr = q0 + r + hi * 8;
        const int kr = kvn0 + l16;
        s[nh][r] = (kr <= qr) ? sa[r] * scale : -1e30f;  // causal mask
      }
    }

    // online softmax over this 16x32 block (row = across 16-lane half)
#pragma unroll
    for (int r = 0; r < 8; r++) {
      float mx = fmaxf(s[0][r], s[1][r]);
#pragma unroll
      for (int off = 8; off >= 1; off >>= 1) mx = fmaxf(mx, __shfl_xor(mx, off));
      const float nm   = fmaxf(rowmax[r], mx);
      const float corr = __expf(rowmax[r] - nm);
      float psum = 0.0f;
#pragma unroll
      for (int nh = 0; nh < 2; nh++) {
        const float p = __expf(s[nh][r] - nm);
        ldsP[(r + hi * 8) * 32 + nh * 16 + l16] = f2bf(p);
        psum += p;
      }
#pragma unroll
      for (int off = 8; off >= 1; off >>= 1) psum += __shfl_xor(psum, off);
      rowsum[r] = rowsum[r] * corr + psum;
      rowmax[r] = nm;
#pragma unroll
      for (int nt = 0; nt < 4; nt++) o[nt][r] *= corr;
    }
    __syncthreads();  // single-wave WG: orders the ldsP store->load

    Frag16 pa;  // P as A-fragment (16x32)
    {
      const unsigned short* p = ldsP + l16 * 32;
      pa.u[0] = *(const B16*)(p + hi * 16);
      pa.u[1] = *(const B16*)(p + hi * 16 + 8);
    }
#pragma unroll
    for (int nt = 0; nt < 4; nt++) {  // O[:, nt*16:+16] += P x V
      // B-frag of V: lane = channel (row of VT), kv contiguous
      Frag16 vb;
      const unsigned short* p = VT + (size_t)(nt * 16 + l16) * BT + kv0 + hi * 16;
      vb.u[0] = *(const B16*)p;
      vb.u[1] = *(const B16*)(p + 8);
      o[nt] = wmma_bf16(pa.v, vb.v, o[nt]);
    }
    __syncthreads();
  }

  // normalize + write concat-head output [B,T,C] bf16
#pragma unroll
  for (int r = 0; r < 8; r++) {
    const float inv = 1.0f / rowsum[r];
    const int t = q0 + r + hi * 8;
#pragma unroll
    for (int nt = 0; nt < 4; nt++) {
      const int n = head * HS + nt * 16 + l16;
      attnout[((size_t)(b * BT + t)) * CD + n] = f2bf(o[nt][r] * inv);
    }
  }
}

// ---------------- host-side orchestration ----------------
extern "C" void kernel_launch(void* const* d_in, const int* in_sizes, int n_in,
                              void* d_out, int out_size, void* d_ws, size_t ws_size,
                              hipStream_t stream) {
  const float* x   = (const float*)d_in[0];
  const float* Wq  = (const float*)d_in[1];
  const float* Wk  = (const float*)d_in[2];
  const float* Wv  = (const float*)d_in[3];
  const float* Wo  = (const float*)d_in[4];
  const float* bo  = (const float*)d_in[5];
  const float* W1  = (const float*)d_in[6];
  const float* b1  = (const float*)d_in[7];
  const float* W2  = (const float*)d_in[8];
  const float* b2  = (const float*)d_in[9];
  const float* g1  = (const float*)d_in[10];
  const float* be1 = (const float*)d_in[11];
  const float* g2  = (const float*)d_in[12];
  const float* be2 = (const float*)d_in[13];
  float* out = (float*)d_out;

  // workspace layout (bytes)
  char* ws = (char*)d_ws;
  size_t off = 0;
  auto alloc = [&](size_t bytes) { char* p = ws + off; off += (bytes + 255) & ~(size_t)255; return p; };
  unsigned short* hbf  = (unsigned short*)alloc((size_t)MROWS * CD * 2);
  unsigned short* qb   = (unsigned short*)alloc((size_t)MROWS * CD * 2);
  unsigned short* kbuf = (unsigned short*)alloc((size_t)MROWS * CD * 2);
  unsigned short* vT   = (unsigned short*)alloc((size_t)MROWS * CD * 2);  // [B,H,hs,T]
  unsigned short* atb  = (unsigned short*)alloc((size_t)MROWS * CD * 2);
  float*          x2   = (float*)        alloc((size_t)MROWS * CD * 4);
  unsigned short* h2   = (unsigned short*)alloc((size_t)MROWS * CD * 2);
  unsigned short* act  = (unsigned short*)alloc((size_t)MROWS * FF * 2);
  unsigned short* wqT  = (unsigned short*)alloc((size_t)NH * CD * HS * 2);  // [H,hs,C]
  unsigned short* wkT  = (unsigned short*)alloc((size_t)NH * CD * HS * 2);
  unsigned short* wvT  = (unsigned short*)alloc((size_t)NH * CD * HS * 2);
  unsigned short* woT  = (unsigned short*)alloc((size_t)CD * CD * 2);      // [C,C]^T
  unsigned short* w1T  = (unsigned short*)alloc((size_t)CD * FF * 2);      // [4C,C]
  unsigned short* w2T  = (unsigned short*)alloc((size_t)FF * CD * 2);      // [C,4C]

  // transpose-convert all weights to bf16 [N,K]
  cvtT_bf16_kernel<<<dim3(CD / 32, HS / 32, NH), 256, 0, stream>>>(Wq, wqT, CD, HS);
  cvtT_bf16_kernel<<<dim3(CD / 32, HS / 32, NH), 256, 0, stream>>>(Wk, wkT, CD, HS);
  cvtT_bf16_kernel<<<dim3(CD / 32, HS / 32, NH), 256, 0, stream>>>(Wv, wvT, CD, HS);
  cvtT_bf16_kernel<<<dim3(CD / 32, CD / 32, 1), 256, 0, stream>>>(Wo, woT, CD, CD);
  cvtT_bf16_kernel<<<dim3(CD / 32, FF / 32, 1), 256, 0, stream>>>(W1, w1T, CD, FF);
  cvtT_bf16_kernel<<<dim3(FF / 32, CD / 32, 1), 256, 0, stream>>>(W2, w2T, FF, CD);

  // LN1
  ln_kernel<<<MROWS, 256, 0, stream>>>(x, g1, be1, hbf);

  // QKV projections: per head GEMM, M=8192 N=64 K=1024 (z = head)
  gemm_bf16_kernel<false, false, false, 2>
      <<<dim3(MROWS / 128, 1, NH), 256, 0, stream>>>(hbf, wqT, nullptr, nullptr,
                                                     nullptr, qb, MROWS, HS, CD);
  gemm_bf16_kernel<false, false, false, 2>
      <<<dim3(MROWS / 128, 1, NH), 256, 0, stream>>>(hbf, wkT, nullptr, nullptr,
                                                     nullptr, kbuf, MROWS, HS, CD);
  gemm_bf16_kernel<false, false, false, 3>
      <<<dim3(MROWS / 128, 1, NH), 256, 0, stream>>>(hbf, wvT, nullptr, nullptr,
                                                     nullptr, vT, MROWS, HS, CD);

  // flash attention
  attn_kernel<<<dim3(BT / 16, NH, NB), 32, 0, stream>>>(qb, kbuf, vT, atb);

  // x2 = x + attn @ Wo + bo
  gemm_bf16_kernel<true, false, true, 0>
      <<<dim3(MROWS / 128, CD / 64, 1), 256, 0, stream>>>(atb, woT, bo, x, x2,
                                                          nullptr, MROWS, CD, CD);
  // LN2
  ln_kernel<<<MROWS, 256, 0, stream>>>(x2, g2, be2, h2);

  // act = relu(h2 @ W1 + b1)
  gemm_bf16_kernel<true, true, false, 1>
      <<<dim3(MROWS / 128, FF / 64, 1), 256, 0, stream>>>(h2, w1T, b1, nullptr,
                                                          nullptr, act, MROWS, FF, CD);
  // out = x2 + act @ W2 + b2
  gemm_bf16_kernel<true, false, true, 0>
      <<<dim3(MROWS / 128, CD / 64, 1), 256, 0, stream>>>(act, w2T, b2, x2, out,
                                                          nullptr, MROWS, CD, FF);
}
